// Net2D_64914135712185
// MI455X (gfx1250) — matrix operand order repaired and checked by reading, not compiled, not run on hardware
//
#include <hip/hip_runtime.h>
#include <hip/hip_bf16.h>

// ---------------------------------------------------------------------------
// CDNA5 (gfx1250) wave32 WMMA types
// ---------------------------------------------------------------------------
typedef __attribute__((ext_vector_type(16))) _Float16     v16h;
typedef __attribute__((ext_vector_type(8)))  float        v8f;
typedef __attribute__((ext_vector_type(4)))  unsigned int u32x4;

union Frag16 {  // 16 halves == 8 VGPRs per lane, loadable as 2x b128
    u32x4 u[2];
    v16h  v;
};

// ---------------------------------------------------------------------------
// Weight repack: HWIO f32 -> f16 WMMA B-fragments.
// Blob layout: [tap*KC + kc][nt][lane(32)][e(16)] halves; a block's two
// consecutive N-tiles are one contiguous 2KB run per K-step (async-DMA friendly).
// Mapping (contiguous k, same order as the A loads): lane 0..15 hold K=kc*32+e, N=nt*16+lane;
// lane 16..31 hold K=kc*32+16+e, N=nt*16+(lane-16).
// ---------------------------------------------------------------------------
__global__ __launch_bounds__(256) void repack_weights(
    const float* __restrict__ w, _Float16* __restrict__ frag,
    int taps, int Cin, int Cinp, int Cout)
{
    const int KC = Cinp >> 5, NT = Cout >> 4;
    const long total = (long)taps * KC * NT * 512;
    long idx = (long)blockIdx.x * blockDim.x + threadIdx.x;
    if (idx >= total) return;
    int e    = (int)(idx & 15);
    int lane = (int)((idx >> 4) & 31);
    long g   = idx >> 9;                 // (t*KC + kc)*NT + nt
    int nt = (int)(g % NT); g /= NT;
    int kc = (int)(g % KC);
    int t  = (int)(g / KC);
    int k  = kc * 32 + (lane >> 4) * 16 + e;   // input channel within tap
    int co = nt * 16 + (lane & 15);
    float v = (k < Cin) ? w[((long)t * Cin + k) * Cout + co] : 0.f;
    frag[idx] = (_Float16)v;
}

// ---------------------------------------------------------------------------
// Input prep: x [B,2,H,W] f32 NCHW * mask -> f16 NHWC, channels padded to 32.
// ---------------------------------------------------------------------------
__global__ __launch_bounds__(256) void prep_input(
    const float* __restrict__ x, const float* __restrict__ mask,
    _Float16* __restrict__ act, int lgH, int lgW, int Cin, int lgCp)
{
    long total = (long)2 << (lgH + lgW + lgCp);
    long idx = (long)blockIdx.x * blockDim.x + threadIdx.x;
    if (idx >= total) return;
    int  c    = (int)(idx & ((1 << lgCp) - 1));
    long site = idx >> lgCp;
    int ww = (int)(site & ((1 << lgW) - 1));
    int hh = (int)((site >> lgW) & ((1 << lgH) - 1));
    int bb = (int)(site >> (lgW + lgH));
    float v = 0.f;
    if (c < Cin) {
        long xi = ((((long)bb * Cin + c) << lgH) + hh);
        xi = (xi << lgW) + ww;
        v = x[xi] * mask[site];
    }
    act[idx] = (_Float16)v;
}

// ---------------------------------------------------------------------------
// Implicit-GEMM submanifold conv, V_WMMA_F32_16X16X32_F16, 2x2 register tile:
//   wave  = 32 output sites x 32 couts (4 f32 accumulators)
//   block = 8 waves, all sharing one 32-cout column (blockIdx.y)
// B-fragments are staged global->LDS with GLOBAL_LOAD_ASYNC_TO_LDS_B128 into a
// 4-slot ring, issued 2 stages ahead; S_WAIT_ASYNCCNT 1 (per-wave ASYNCcnt sees
// one issue per stage, so "<=1 outstanding" == "my next stage has landed") +
// workgroup barrier. Each DMA gets a full 4-WMMA iteration to hide its latency.
// A-fragments load directly in contiguous k order, k = 16*hi + e, matching the B repack (two 16B runs).
// ---------------------------------------------------------------------------
template<int KSIZE>
__global__ __launch_bounds__(256) void conv_wmma(
    const _Float16* __restrict__ act, int inCStride, int chanOff, int KC,
    const _Float16* __restrict__ wfrag, int NT, int Cout,
    const float* __restrict__ mask, float* __restrict__ out,
    int lgH, int lgW, int S)
{
    __shared__ __align__(16) _Float16 sbuf[4][1024];   // 4-slot ring x 2KB

    const int tid  = threadIdx.x;
    const int lane = tid & 31;
    const int wave = tid >> 5;
    const int Mtiles = S >> 5;
    int mt = blockIdx.x * 8 + wave;
    if (mt >= Mtiles) mt = Mtiles - 1;        // stay for barriers; dup stores benign
    const int nt0 = blockIdx.y * 2;
    const int m  = lane & 15;
    const int hi = lane >> 4;
    const int H = 1 << lgH, W = 1 << lgW;
    constexpr int taps = KSIZE * KSIZE, pad = KSIZE / 2;
    const int ST = taps * KC;                  // >= 9 for all layers

    // the two A-row sites this lane covers
    int site0 = mt * 32 + m;
    int site1 = site0 + 16;
    int w0 = site0 & (W - 1), h0 = (site0 >> lgW) & (H - 1), b0 = site0 >> (lgW + lgH);
    int w1 = site1 & (W - 1), h1 = (site1 >> lgW) & (H - 1), b1 = site1 >> (lgW + lgH);

    // stage s: 2KB of B-fragments (both N-tiles); waves 0..3 move 16B per lane
    auto issue = [&](int s) {
        if (tid < 128) {
            const _Float16* src = wfrag + ((size_t)s * NT + nt0) * 512 + (size_t)tid * 8;
            unsigned dst = (unsigned)(size_t)&sbuf[s & 3][tid * 8];
            asm volatile("global_load_async_to_lds_b128 %0, %1, off"
                         :: "v"(dst), "v"((unsigned long long)(size_t)src)
                         : "memory");
        }
    };
    auto flush1 = [&]() {                       // next stage landed (<=1 in flight)
        if (tid < 128)
            asm volatile("s_wait_asynccnt 1" ::: "memory");
        __syncthreads();
    };

    issue(0);
    issue(1);
    if (tid < 128)
        asm volatile("s_wait_asynccnt 1" ::: "memory");   // stage 0 landed
    __syncthreads();

    v8f acc00 = {0.f,0.f,0.f,0.f,0.f,0.f,0.f,0.f};
    v8f acc01 = acc00, acc10 = acc00, acc11 = acc00;

    int s = 0;
    #pragma unroll
    for (int t = 0; t < taps; ++t) {
        const int dh = t / KSIZE - pad, dw = t % KSIZE - pad;  // constants (unrolled)
        int hh0 = h0 + dh, ww0 = w0 + dw;
        int hh1 = h1 + dh, ww1 = w1 + dw;
        bool v0 = ((unsigned)hh0 < (unsigned)H) & ((unsigned)ww0 < (unsigned)W);
        bool v1 = ((unsigned)hh1 < (unsigned)H) & ((unsigned)ww1 < (unsigned)W);
        const _Float16* ab0 = act +
            ((((size_t)(b0 << lgH) + hh0) << lgW) + ww0) * inCStride + chanOff + hi * 16;
        const _Float16* ab1 = act +
            ((((size_t)(b1 << lgH) + hh1) << lgW) + ww1) * inCStride + chanOff + hi * 16;

        for (int kc = 0; kc < KC; ++kc, ++s) {
            if (s + 2 < ST) issue(s + 2);       // ring slot (s+2)&3, consumed 2 ago

            u32x4 z = 0;
            Frag16 A0, A1;
            A0.u[0] = z; A0.u[1] = z; A1.u[0] = z; A1.u[1] = z;
            if (v0) {
                const u32x4* p = (const u32x4*)(ab0 + kc * 32);
                A0.u[0] = p[0];            // K = kc*32 + hi*16 + 0..7
                A0.u[1] = p[1];            // K = kc*32 + hi*16 + 8..15
            }
            if (v1) {
                const u32x4* p = (const u32x4*)(ab1 + kc * 32);
                A1.u[0] = p[0];
                A1.u[1] = p[1];
            }
            // B from LDS stage: frag j at bytes j*1024 + lane*32
            const u32x4* bp = (const u32x4*)&sbuf[s & 3][0];
            Frag16 B0, B1;
            B0.u[0] = bp[lane * 2];
            B0.u[1] = bp[lane * 2 + 1];
            B1.u[0] = bp[64 + lane * 2];
            B1.u[1] = bp[64 + lane * 2 + 1];

            acc00 = __builtin_amdgcn_wmma_f32_16x16x32_f16(false, A0.v, false, B0.v, (short)0, acc00, false, false);
            acc01 = __builtin_amdgcn_wmma_f32_16x16x32_f16(false, A0.v, false, B1.v, (short)0, acc01, false, false);
            acc10 = __builtin_amdgcn_wmma_f32_16x16x32_f16(false, A1.v, false, B0.v, (short)0, acc10, false, false);
            acc11 = __builtin_amdgcn_wmma_f32_16x16x32_f16(false, A1.v, false, B1.v, (short)0, acc11, false, false);

            flush1();                           // stage s+1 is now resident
        }
    }

    // masked store; C/D layout: lane = N + 16*(M>=8), vgpr r = M&7
    const int n = lane & 15;
    #pragma unroll
    for (int r = 0; r < 8; ++r) {
        int s0 = mt * 32 + hi * 8 + r;
        int s1 = s0 + 16;
        float mv0 = mask ? mask[s0] : 1.0f;
        float mv1 = mask ? mask[s1] : 1.0f;
        out[(size_t)s0 * Cout + (nt0 + 0) * 16 + n] = acc00[r] * mv0;
        out[(size_t)s0 * Cout + (nt0 + 1) * 16 + n] = acc01[r] * mv0;
        out[(size_t)s1 * Cout + (nt0 + 0) * 16 + n] = acc10[r] * mv1;
        out[(size_t)s1 * Cout + (nt0 + 1) * 16 + n] = acc11[r] * mv1;
    }
}

// ---------------------------------------------------------------------------
// Per-channel sum / sum-of-squares over NHWC f32 (deterministic tree reduce).
// ---------------------------------------------------------------------------
__global__ __launch_bounds__(256) void bn_stats(
    const float* __restrict__ conv, int S, int lgC, float* __restrict__ stats)
{
    int c = blockIdx.x;
    float s = 0.f, sq = 0.f;
    for (int i = threadIdx.x; i < S; i += 256) {
        float v = conv[((long)i << lgC) + c];
        s += v;
        sq += v * v;
    }
    __shared__ float sh0[256];
    __shared__ float sh1[256];
    sh0[threadIdx.x] = s;
    sh1[threadIdx.x] = sq;
    __syncthreads();
    for (int o = 128; o > 0; o >>= 1) {
        if ((int)threadIdx.x < o) {
            sh0[threadIdx.x] += sh0[threadIdx.x + o];
            sh1[threadIdx.x] += sh1[threadIdx.x + o];
        }
        __syncthreads();
    }
    if (threadIdx.x == 0) {
        stats[2 * c]     = sh0[0];
        stats[2 * c + 1] = sh1[0];
    }
}

__global__ __launch_bounds__(256) void reduce_mask(
    const float* __restrict__ mask, int S, float* __restrict__ out)
{
    float s = 0.f;
    for (int i = threadIdx.x; i < S; i += 256) s += mask[i];
    __shared__ float sh[256];
    sh[threadIdx.x] = s;
    __syncthreads();
    for (int o = 128; o > 0; o >>= 1) {
        if ((int)threadIdx.x < o) sh[threadIdx.x] += sh[threadIdx.x + o];
        __syncthreads();
    }
    if (threadIdx.x == 0) out[0] = sh[0];
}

__global__ void fill_ones(float* p, int n)
{
    int i = blockIdx.x * blockDim.x + threadIdx.x;
    if (i < n) p[i] = 1.0f;
}

// ---------------------------------------------------------------------------
// BN (active-site stats) + LeakyReLU + mask, fused with mask-aware maxpool.
// ---------------------------------------------------------------------------
__global__ __launch_bounds__(256) void bn_apply_pool(
    const float* __restrict__ conv, const float* __restrict__ stats,
    const float* __restrict__ gamma, const float* __restrict__ beta,
    const float* __restrict__ maskIn, const float* __restrict__ countPtr,
    int lgH, int lgW, int lgC, int ph, int pw,
    _Float16* __restrict__ outAct, int outCStride, int outChanOff,
    float* __restrict__ outMask, float leak, float eps)
{
    const int lgPh = (ph == 2), lgPw = (pw == 2);
    const int lgHo = lgH - lgPh, lgWo = lgW - lgPw;
    long total = (long)2 << (lgHo + lgWo + lgC);
    long idx = (long)blockIdx.x * blockDim.x + threadIdx.x;
    if (idx >= total) return;
    int c  = (int)(idx & ((1 << lgC) - 1));
    long t2 = idx >> lgC;
    int wo = (int)(t2 & ((1 << lgWo) - 1)); t2 >>= lgWo;
    int ho = (int)(t2 & ((1 << lgHo) - 1));
    int b  = (int)(t2 >> lgHo);

    float n = countPtr[0];
    if (n < 1.f) n = 1.f;
    float mean  = stats[2 * c] / n;
    float var   = stats[2 * c + 1] / n - mean * mean;
    float scale = rsqrtf(var + eps) * gamma[c];
    float shift = beta[c];

    float best = -__builtin_inff();
    float pm = 0.f;
    for (int i = 0; i < ph; ++i) {
        for (int j = 0; j < pw; ++j) {
            long sidx = ((((long)b << lgH) + (ho << lgPh) + i) << lgW) + (wo << lgPw) + j;
            float mv = maskIn[sidx];
            float v  = conv[(sidx << lgC) + c];
            float xn = (v - mean) * scale + shift;
            float l  = (xn >= 0.f) ? xn : leak * xn;
            float cand = (mv > 0.f) ? (l * mv) : -__builtin_inff();
            best = fmaxf(best, cand);
            pm   = fmaxf(pm, mv);
        }
    }
    float res = (pm > 0.f) ? best : 0.f;
    long osite = ((((long)b << lgHo) + ho) << lgWo) + wo;
    outAct[osite * outCStride + outChanOff + c] = (_Float16)res;
    if (c == 0 && outMask) outMask[osite] = pm;
}

// ---------------------------------------------------------------------------
// Final 1x1 conv 1024->4 + bias, output NCHW [B,4,8,8] (tiny: plain dots).
// ---------------------------------------------------------------------------
__global__ __launch_bounds__(256) void last_conv(
    const _Float16* __restrict__ y, const float* __restrict__ lw,
    const float* __restrict__ lb, float* __restrict__ out, int HW, int C)
{
    int idx = blockIdx.x * blockDim.x + threadIdx.x;  // B*HW*4
    if (idx >= 2 * HW * 4) return;
    int oc = idx & 3;
    int site = idx >> 2;
    float acc = lb[oc];
    for (int c = 0; c < C; ++c)
        acc += (float)y[(long)site * C + c] * lw[c * 4 + oc];
    int b  = site / HW;
    int hw = site % HW;
    out[((long)b * 4 + oc) * HW + hw] = acc;
}

// ---------------------------------------------------------------------------
// Driver
// ---------------------------------------------------------------------------
static inline long cdiv(long a, long b) { return (a + b - 1) / b; }

extern "C" void kernel_launch(void* const* d_in, const int* in_sizes, int n_in,
                              void* d_out, int out_size, void* d_ws, size_t ws_size,
                              hipStream_t stream)
{
    (void)in_sizes; (void)n_in; (void)out_size; (void)ws_size;

    const float* x     = (const float*)d_in[0];
    const float* mask0 = (const float*)d_in[1];
    const float* Wt[10]; const float* Gm[10]; const float* Bt[10];
    for (int i = 0; i < 10; ++i) {
        Wt[i] = (const float*)d_in[2 + 3 * i];
        Gm[i] = (const float*)d_in[3 + 3 * i];
        Bt[i] = (const float*)d_in[4 + 3 * i];
    }
    const float* comb_w = (const float*)d_in[32];
    const float* comb_g = (const float*)d_in[33];
    const float* comb_b = (const float*)d_in[34];
    const float* lw     = (const float*)d_in[35];
    const float* lb     = (const float*)d_in[36];
    float* out = (float*)d_out;

    // ---- workspace carve ----
    char* ws = (char*)d_ws;
    size_t off = 0;
    auto carve = [&](size_t bytes) -> void* {
        void* p = ws + off;
        off = (off + bytes + 255) & ~(size_t)255;
        return p;
    };
    float*     convOut = (float*)carve(33554432);             // max 2*128*1024*32*4
    _Float16*  actA    = (_Float16*)carve(16777216);
    _Float16*  actB    = (_Float16*)carve(16777216);
    _Float16*  concat  = (_Float16*)carve((size_t)128 * 2048 * 2);
    _Float16*  yAct    = (_Float16*)carve((size_t)128 * 1024 * 2);
    const int  mS[8] = {262144, 131072, 65536, 32768, 8192, 2048, 512, 128};
    const float* mP[8];
    mP[0] = mask0;
    float* mBuf[8];
    mBuf[0] = nullptr;
    for (int i = 1; i < 8; ++i) { mBuf[i] = (float*)carve((size_t)mS[i] * 4); mP[i] = mBuf[i]; }
    float* ones   = (float*)carve(512);
    float* counts = (float*)carve(256);      // [0..7] mask levels, [8] head
    float* stats  = (float*)carve(8192);     // 2*1024 f32

    // ---- layer table ----
    const int Cin [10] = {2, 32, 64, 128, 256, 512, 256, 512, 1024, 1024};
    const int Cinp[10] = {32, 32, 64, 128, 256, 512, 256, 512, 1024, 1024};
    const int Cout[10] = {32, 64, 128, 256, 512, 256, 512, 1024, 1024, 1024};
    const int lgCo[10] = {5, 6, 7, 8, 9, 8, 9, 10, 10, 10};
    const int Ks  [10] = {3, 3, 3, 3, 3, 1, 3, 3, 3, 3};
    const int lgHh[10] = {7, 7, 7, 7, 6, 5, 5, 4, 3, 3};
    const int lgWw[10] = {10, 9, 8, 7, 6, 5, 5, 4, 3, 3};
    const int Ph  [10] = {1, 1, 1, 2, 2, 1, 2, 2, 1, 1};
    const int Pw  [10] = {2, 2, 2, 2, 2, 1, 2, 2, 1, 1};
    const int mIdx[10] = {0, 1, 2, 3, 4, 5, 5, 6, 7, 7};     // mask level used
    const int pOut[10] = {1, 2, 3, 4, 5, -1, 6, 7, -1, -1};  // mask level produced

    const _Float16* inA [10] = {actA, actB, actA, actB, actA, actB, actA, actB,
                                concat, actA};
    const int inStride  [10] = {32, 32, 64, 128, 256, 512, 256, 512, 2048, 1024};
    const int inOff     [10] = {0, 0, 0, 0, 0, 0, 0, 0, 1024, 0};
    _Float16* outA      [10] = {actB, actA, actB, actA, actB, actA, actB, concat,
                                actA, concat};
    const int outStride [10] = {32, 64, 128, 256, 512, 256, 512, 2048, 1024, 2048};
    const int outOff    [10] = {0, 0, 0, 0, 0, 0, 0, 1024, 0, 0};

    _Float16* frag[11];
    for (int l = 0; l < 10; ++l)
        frag[l] = (_Float16*)carve((size_t)Ks[l] * Ks[l] * Cinp[l] * Cout[l] * 2);
    frag[10] = (_Float16*)carve((size_t)2048 * 1024 * 2);    // comb 1x1

    // ---- prologue ----
    fill_ones<<<1, 256, 0, stream>>>(ones, 128);
    reduce_mask<<<1, 256, 0, stream>>>(mask0, mS[0], counts + 0);
    reduce_mask<<<1, 256, 0, stream>>>(ones, 128, counts + 8);
    {
        long total = (long)2 * 128 * 1024 * 32;
        prep_input<<<dim3(cdiv(total, 256)), 256, 0, stream>>>(
            x, mask0, actA, 7, 10, 2, 5);
    }

    auto launchConv = [&](const _Float16* in, int inStr, int chOff, int Cinp_,
                          const _Float16* fr, int Cout_, const float* mk,
                          float* cOut, int lgH_, int lgW_, int ksize) {
        int S_ = 2 << (lgH_ + lgW_);
        dim3 g((unsigned)cdiv(S_ / 32, 8), (unsigned)(Cout_ / 32));
        if (ksize == 3)
            conv_wmma<3><<<g, 256, 0, stream>>>(in, inStr, chOff, Cinp_ >> 5,
                fr, Cout_ >> 4, Cout_, mk, cOut, lgH_, lgW_, S_);
        else
            conv_wmma<1><<<g, 256, 0, stream>>>(in, inStr, chOff, Cinp_ >> 5,
                fr, Cout_ >> 4, Cout_, mk, cOut, lgH_, lgW_, S_);
    };

    // ---- sparse stack ----
    for (int l = 0; l < 10; ++l) {
        const int S = 2 << (lgHh[l] + lgWw[l]);
        const int taps = Ks[l] * Ks[l];

        long rtot = (long)taps * Cinp[l] * Cout[l];
        repack_weights<<<dim3(cdiv(rtot, 256)), 256, 0, stream>>>(
            Wt[l], frag[l], taps, Cin[l], Cinp[l], Cout[l]);

        launchConv(inA[l], inStride[l], inOff[l], Cinp[l], frag[l], Cout[l],
                   mP[mIdx[l]], convOut, lgHh[l], lgWw[l], Ks[l]);

        bn_stats<<<Cout[l], 256, 0, stream>>>(convOut, S, lgCo[l], stats);

        long atot = (long)S / (Ph[l] * Pw[l]) * Cout[l];
        bn_apply_pool<<<dim3(cdiv(atot, 256)), 256, 0, stream>>>(
            convOut, stats, Gm[l], Bt[l], mP[mIdx[l]], counts + mIdx[l],
            lgHh[l], lgWw[l], lgCo[l], Ph[l], Pw[l],
            outA[l], outStride[l], outOff[l],
            (pOut[l] >= 0) ? mBuf[pOut[l]] : nullptr, 0.333f, 1e-4f);

        if (pOut[l] >= 0)
            reduce_mask<<<1, 256, 0, stream>>>(
                mP[pOut[l]], mS[pOut[l]], counts + pOut[l]);
    }

    // ---- dense head ----
    repack_weights<<<dim3(cdiv((long)2048 * 1024, 256)), 256, 0, stream>>>(
        comb_w, frag[10], 1, 2048, 2048, 1024);
    launchConv(concat, 2048, 0, 2048, frag[10], 1024, nullptr, convOut, 3, 3, 1);
    bn_stats<<<1024, 256, 0, stream>>>(convOut, 128, 10, stats);
    bn_apply_pool<<<dim3(cdiv((long)128 * 1024, 256)), 256, 0, stream>>>(
        convOut, stats, comb_g, comb_b, ones, counts + 8,
        3, 3, 10, 1, 1, yAct, 1024, 0, nullptr, 0.01f, 1e-5f);

    last_conv<<<2, 256, 0, stream>>>(yAct, lw, lb, out, 64, 1024);
}
